// MHA_28063316312438
// MI455X (gfx1250) — compile-verified, matrix operand order checked by reading, not run on hardware
//
#include <hip/hip_runtime.h>

// MHA (cross-attention) for MI455X / gfx1250, wave32 + WMMA f16->f32.
// d_out = [out (2*2048*1024 f32)] ++ [attn (2*16*2048*2048 f32)]

#define E_DIM 1024
#define HEADS 16
#define HDIM  64
#define SEQ   2048
#define BATCH 2
#define NROWS 4096  // BATCH * SEQ

typedef __attribute__((ext_vector_type(16))) _Float16 v16h;
typedef __attribute__((ext_vector_type(4)))  _Float16 v4h;
typedef __attribute__((ext_vector_type(8)))  float    v8f;
typedef __attribute__((ext_vector_type(4)))  float    v4f;

// A/B 16-bit fragment: lane holds 16 halves = two contiguous runs of 8 along K:
//   K = e + 8*(e>>3) + 8*(lane>>4)  (ISA 7.12.2 16-bit A 16x32 layout)
__device__ __forceinline__ v16h frag_from_f16(const _Float16* p, int hf) {
  v16h r;
  const _Float16* p0 = p + 8 * hf;
  const _Float16* p1 = p0 + 16;
#pragma unroll
  for (int i = 0; i < 8; ++i) { r[i] = p0[i]; r[i + 8] = p1[i]; }
  return r;
}

__device__ __forceinline__ v8f wmma_f16(v16h a, v16h b, v8f c) {
  return __builtin_amdgcn_wmma_f32_16x16x32_f16(false, a, false, b, (short)0, c,
                                                false, false);
}

// ---------------------------------------------------------------------------
// Kernel 0: elementwise f32 -> f16 (vectorized x4). n must be multiple of 4.
// ---------------------------------------------------------------------------
__global__ __launch_bounds__(256) void cvt_f16_kernel(
    const float* __restrict__ in, _Float16* __restrict__ out, int n) {
  const int i = (blockIdx.x * blockDim.x + threadIdx.x) * 4;
  if (i < n) {
    const float4 v = *(const float4*)(in + i);
    v4h o;
    o[0] = (_Float16)v.x; o[1] = (_Float16)v.y;
    o[2] = (_Float16)v.z; o[3] = (_Float16)v.w;
    *(v4h*)(out + i) = o;
  }
}

// ---------------------------------------------------------------------------
// Kernel 1: fused QKV projections (all-f16 operands, f32 accumulate).
// z=0: Q = tgt@Wq^T (scaled 1/sqrt(D)), z=1: K = src@Wk^T,
// z=2: V = src@Wv^T stored transposed as Vt[b,h,d,s].
// Block = 256 threads (8 waves); wave computes a 32x64 tile (8 accums).
// ---------------------------------------------------------------------------
__global__ __launch_bounds__(256) void qkv_proj_kernel(
    const _Float16* __restrict__ srcH, const _Float16* __restrict__ tgtH,
    const _Float16* __restrict__ WH,   // [Wq|Wk|Wv|Wo] f16, each E*E
    _Float16* __restrict__ Qf, _Float16* __restrict__ Kf,
    _Float16* __restrict__ Vt) {
  const int lane = threadIdx.x & 31;
  const int hf   = lane >> 4;
  const int lm   = lane & 15;
  const int wid  = threadIdx.x >> 5;
  const int mode = blockIdx.z;
  const int rowBase = blockIdx.y * 256 + wid * 32;  // M tile (32 rows)
  const int colBase = blockIdx.x * 64;              // N tile (64 cols)

  const _Float16* A = (mode == 0) ? tgtH : srcH;
  const _Float16* W = WH + (size_t)mode * E_DIM * E_DIM;

  const _Float16* Ar0 = A + (size_t)(rowBase + lm) * E_DIM;
  const _Float16* Ar1 = A + (size_t)(rowBase + 16 + lm) * E_DIM;
  const _Float16* Wr0 = W + (size_t)(colBase +  0 + lm) * E_DIM;
  const _Float16* Wr1 = W + (size_t)(colBase + 16 + lm) * E_DIM;
  const _Float16* Wr2 = W + (size_t)(colBase + 32 + lm) * E_DIM;
  const _Float16* Wr3 = W + (size_t)(colBase + 48 + lm) * E_DIM;

  v8f c00 = {}, c01 = {}, c02 = {}, c03 = {};
  v8f c10 = {}, c11 = {}, c12 = {}, c13 = {};
  for (int kc = 0; kc < E_DIM; kc += 32) {
    if (kc + 128 < E_DIM) {
      __builtin_prefetch(Ar0 + kc + 128, 0, 1);
      __builtin_prefetch(Ar1 + kc + 128, 0, 1);
    }
    v16h a0 = frag_from_f16(Ar0 + kc, hf);
    v16h a1 = frag_from_f16(Ar1 + kc, hf);
    v16h b0 = frag_from_f16(Wr0 + kc, hf);
    v16h b1 = frag_from_f16(Wr1 + kc, hf);
    v16h b2 = frag_from_f16(Wr2 + kc, hf);
    v16h b3 = frag_from_f16(Wr3 + kc, hf);
    c00 = wmma_f16(a0, b0, c00);
    c01 = wmma_f16(a0, b1, c01);
    c02 = wmma_f16(a0, b2, c02);
    c03 = wmma_f16(a0, b3, c03);
    c10 = wmma_f16(a1, b0, c10);
    c11 = wmma_f16(a1, b1, c11);
    c12 = wmma_f16(a1, b2, c12);
    c13 = wmma_f16(a1, b3, c13);
  }

  v8f accs[2][4] = {{c00, c01, c02, c03}, {c10, c11, c12, c13}};
#pragma unroll
  for (int ms = 0; ms < 2; ++ms) {
#pragma unroll
    for (int nt = 0; nt < 4; ++nt) {
#pragma unroll
      for (int r = 0; r < 8; ++r) {
        const int gr = rowBase + ms * 16 + r + 8 * hf;  // row = b*SEQ + t|s
        const int gc = colBase + nt * 16 + lm;          // col = h*HDIM + d
        const float v = accs[ms][nt][r];
        if (mode == 0) {
          Qf[(size_t)gr * E_DIM + gc] = (_Float16)(v * 0.125f);  // 1/sqrt(64)
        } else if (mode == 1) {
          Kf[(size_t)gr * E_DIM + gc] = (_Float16)v;
        } else {
          const int b = gr >> 11, s = gr & 2047;
          const int h = gc >> 6,  d = gc & 63;
          Vt[(((size_t)(b * HEADS + h) * HDIM + d) << 11) + s] = (_Float16)v;
        }
      }
    }
  }
}

// ---------------------------------------------------------------------------
// Kernel 2: one wave per (b, h, 16-row T tile).
//   scores = Q K^T -> LDS (f16, 16x2048 = 64KB), row softmax (wave32 shfl,
//   x4-vectorized passes), normalized attn streamed as b128 f32 stores,
//   ctx = attn @ V via WMMA from LDS.
// ---------------------------------------------------------------------------
__global__ __launch_bounds__(32) void attn_kernel(
    const _Float16* __restrict__ Qf, const _Float16* __restrict__ Kf,
    const _Float16* __restrict__ Vt, float* __restrict__ attn,
    _Float16* __restrict__ Cx) {
  __shared__ _Float16 sS[16][SEQ];  // 64 KB -> 5 WGs per 320KB WGP
  const int lane = threadIdx.x & 31;
  const int hf   = lane >> 4;
  const int lm   = lane & 15;
  const int t0 = blockIdx.x * 16;
  const int h  = blockIdx.y;
  const int b  = blockIdx.z;

  // Q fragments for this 16-row tile (D = 64 -> two K=32 chunks)
  const _Float16* Qrow = Qf + ((size_t)(b * SEQ + t0 + lm) * E_DIM + h * HDIM);
  const v16h qa0 = frag_from_f16(Qrow, hf);
  const v16h qa1 = frag_from_f16(Qrow + 32, hf);

  // scores: sweep S in 16-col tiles
  for (int st = 0; st < SEQ / 16; ++st) {
    const _Float16* Krow =
        Kf + ((size_t)(b * SEQ + st * 16 + lm) * E_DIM + h * HDIM);
    v16h kb0 = frag_from_f16(Krow, hf);
    v16h kb1 = frag_from_f16(Krow + 32, hf);
    v8f c = {};
    c = wmma_f16(qa0, kb0, c);
    c = wmma_f16(qa1, kb1, c);
#pragma unroll
    for (int r = 0; r < 8; ++r)
      sS[r + 8 * hf][st * 16 + lm] = (_Float16)c[r];
  }
  __syncthreads();  // single-wave WG: compiler fence, barrier -> S_NOP

  // row softmax, x4-vectorized (SEQ/4 = 512 groups, 16 iters/lane/pass)
  for (int r = 0; r < 16; ++r) {
    v4h* row4 = (v4h*)sS[r];

    // pass 1: row max
    float m = -1e30f;
    for (int j = lane; j < SEQ / 4; j += 32) {
      const v4h x = row4[j];
      m = fmaxf(m, fmaxf(fmaxf((float)x[0], (float)x[1]),
                         fmaxf((float)x[2], (float)x[3])));
    }
#pragma unroll
    for (int o = 16; o > 0; o >>= 1) m = fmaxf(m, __shfl_xor(m, o));

    // pass 2: e = exp(x-m) (<=1.0, exact in f16 range), overwrite row, sum
    float s = 0.f;
    for (int j = lane; j < SEQ / 4; j += 32) {
      const v4h x = row4[j];
      const float e0 = __expf((float)x[0] - m);
      const float e1 = __expf((float)x[1] - m);
      const float e2 = __expf((float)x[2] - m);
      const float e3 = __expf((float)x[3] - m);
      s += (e0 + e1) + (e2 + e3);
      v4h e;
      e[0] = (_Float16)e0; e[1] = (_Float16)e1;
      e[2] = (_Float16)e2; e[3] = (_Float16)e3;
      row4[j] = e;
    }
#pragma unroll
    for (int o = 16; o > 0; o >>= 1) s += __shfl_xor(s, o);
    const float inv = 1.0f / s;

    // pass 3: normalize; stream f32 attn row as b128; keep f16 copy in LDS
    v4f* arow4 =
        (v4f*)(attn + ((size_t)(b * HEADS + h) * SEQ + (t0 + r)) * SEQ);
    for (int j = lane; j < SEQ / 4; j += 32) {
      const v4h e = row4[j];
      v4f a;
      a[0] = (float)e[0] * inv; a[1] = (float)e[1] * inv;
      a[2] = (float)e[2] * inv; a[3] = (float)e[3] * inv;
      arow4[j] = a;  // 512 MB stream: the HBM-bound part (global_store_b128)
      v4h an;
      an[0] = (_Float16)a[0]; an[1] = (_Float16)a[1];
      an[2] = (_Float16)a[2]; an[3] = (_Float16)a[3];
      row4[j] = an;
    }
  }
  __syncthreads();

  // ctx = attn @ V  (contract over S, 4 D-tiles of 16)
  v8f cD0 = {}, cD1 = {}, cD2 = {}, cD3 = {};
  for (int sc = 0; sc < SEQ; sc += 32) {
    v16h a;
    const int k0 = sc + 8 * hf;
#pragma unroll
    for (int i = 0; i < 8; ++i) {
      a[i]     = sS[lm][k0 + i];
      a[i + 8] = sS[lm][k0 + 16 + i];
    }
    const _Float16* Vb = Vt + ((size_t)(b * HEADS + h) * HDIM) * SEQ + sc;
    cD0 = wmma_f16(a, frag_from_f16(Vb + (size_t)( 0 + lm) * SEQ, hf), cD0);
    cD1 = wmma_f16(a, frag_from_f16(Vb + (size_t)(16 + lm) * SEQ, hf), cD1);
    cD2 = wmma_f16(a, frag_from_f16(Vb + (size_t)(32 + lm) * SEQ, hf), cD2);
    cD3 = wmma_f16(a, frag_from_f16(Vb + (size_t)(48 + lm) * SEQ, hf), cD3);
  }

  v8f cDs[4] = {cD0, cD1, cD2, cD3};
#pragma unroll
  for (int dt = 0; dt < 4; ++dt)
#pragma unroll
    for (int r = 0; r < 8; ++r)
      Cx[(size_t)(b * SEQ + t0 + r + 8 * hf) * E_DIM + h * HDIM + dt * 16 + lm] =
          (_Float16)cDs[dt][r];
}

// ---------------------------------------------------------------------------
// Kernel 3: out = ctx @ Wo^T + bo (f32 result).  32x64 per wave, all f16 in.
// ---------------------------------------------------------------------------
__global__ __launch_bounds__(256) void out_proj_kernel(
    const _Float16* __restrict__ Cx, const _Float16* __restrict__ WoH,
    const float* __restrict__ bo, float* __restrict__ out) {
  const int lane = threadIdx.x & 31;
  const int hf   = lane >> 4;
  const int lm   = lane & 15;
  const int wid  = threadIdx.x >> 5;
  const int rowBase = blockIdx.y * 256 + wid * 32;
  const int colBase = blockIdx.x * 64;

  const _Float16* Ar0 = Cx + (size_t)(rowBase + lm) * E_DIM;
  const _Float16* Ar1 = Cx + (size_t)(rowBase + 16 + lm) * E_DIM;
  const _Float16* Wr0 = WoH + (size_t)(colBase +  0 + lm) * E_DIM;
  const _Float16* Wr1 = WoH + (size_t)(colBase + 16 + lm) * E_DIM;
  const _Float16* Wr2 = WoH + (size_t)(colBase + 32 + lm) * E_DIM;
  const _Float16* Wr3 = WoH + (size_t)(colBase + 48 + lm) * E_DIM;

  v8f c00 = {}, c01 = {}, c02 = {}, c03 = {};
  v8f c10 = {}, c11 = {}, c12 = {}, c13 = {};
  for (int kc = 0; kc < E_DIM; kc += 32) {
    if (kc + 128 < E_DIM) {
      __builtin_prefetch(Ar0 + kc + 128, 0, 1);
      __builtin_prefetch(Ar1 + kc + 128, 0, 1);
    }
    v16h a0 = frag_from_f16(Ar0 + kc, hf);
    v16h a1 = frag_from_f16(Ar1 + kc, hf);
    v16h b0 = frag_from_f16(Wr0 + kc, hf);
    v16h b1 = frag_from_f16(Wr1 + kc, hf);
    v16h b2 = frag_from_f16(Wr2 + kc, hf);
    v16h b3 = frag_from_f16(Wr3 + kc, hf);
    c00 = wmma_f16(a0, b0, c00);
    c01 = wmma_f16(a0, b1, c01);
    c02 = wmma_f16(a0, b2, c02);
    c03 = wmma_f16(a0, b3, c03);
    c10 = wmma_f16(a1, b0, c10);
    c11 = wmma_f16(a1, b1, c11);
    c12 = wmma_f16(a1, b2, c12);
    c13 = wmma_f16(a1, b3, c13);
  }

  v8f accs[2][4] = {{c00, c01, c02, c03}, {c10, c11, c12, c13}};
#pragma unroll
  for (int ms = 0; ms < 2; ++ms) {
#pragma unroll
    for (int nt = 0; nt < 4; ++nt) {
#pragma unroll
      for (int r = 0; r < 8; ++r) {
        const int gr = rowBase + ms * 16 + r + 8 * hf;
        const int gc = colBase + nt * 16 + lm;
        out[(size_t)gr * E_DIM + gc] = accs[ms][nt][r] + bo[gc];
      }
    }
  }
}

extern "C" void kernel_launch(void* const* d_in, const int* in_sizes, int n_in,
                              void* d_out, int out_size, void* d_ws,
                              size_t ws_size, hipStream_t stream) {
  const float* src = (const float*)d_in[0];  // source_emb [2,2048,1024]
  const float* tgt = (const float*)d_in[1];  // target_emb [2,2048,1024]
  const float* Wq  = (const float*)d_in[2];
  const float* Wk  = (const float*)d_in[3];
  const float* Wv  = (const float*)d_in[4];
  const float* Wo  = (const float*)d_in[5];
  const float* bo  = (const float*)d_in[6];

  float* out  = (float*)d_out;                        // [2,2048,1024]
  float* attn = out + (size_t)BATCH * SEQ * E_DIM;    // [2,16,2048,2048]

  const size_t NE = (size_t)NROWS * E_DIM;            // 4M elements
  const size_t WE = (size_t)E_DIM * E_DIM;            // 1M elements
  _Float16* Qf   = (_Float16*)d_ws;   // [4096,1024] (pre-scaled)
  _Float16* Kf   = Qf + NE;           // [4096,1024]
  _Float16* Vt   = Kf + NE;           // [b,h,d,s]
  _Float16* Cx   = Vt + NE;           // ctx [4096,1024]
  _Float16* tgtH = Cx + NE;           // f16 target_emb
  _Float16* srcH = tgtH + NE;         // f16 source_emb
  _Float16* WH   = srcH + NE;         // f16 [Wq|Wk|Wv|Wo]
  // total scratch: 28M halves = 56 MB

  // 0) convert inputs/weights to f16 once (removes cvt from GEMM hot loops)
  cvt_f16_kernel<<<dim3(NE / 1024), 256, 0, stream>>>(tgt, tgtH, (int)NE);
  cvt_f16_kernel<<<dim3(NE / 1024), 256, 0, stream>>>(src, srcH, (int)NE);
  cvt_f16_kernel<<<dim3(WE / 1024), 256, 0, stream>>>(Wq, WH + 0 * WE, (int)WE);
  cvt_f16_kernel<<<dim3(WE / 1024), 256, 0, stream>>>(Wk, WH + 1 * WE, (int)WE);
  cvt_f16_kernel<<<dim3(WE / 1024), 256, 0, stream>>>(Wv, WH + 2 * WE, (int)WE);
  cvt_f16_kernel<<<dim3(WE / 1024), 256, 0, stream>>>(Wo, WH + 3 * WE, (int)WE);

  // 1) QKV projections
  qkv_proj_kernel<<<dim3(E_DIM / 64, NROWS / 256, 3), 256, 0, stream>>>(
      srcH, tgtH, WH, Qf, Kf, Vt);
  // 2) attention (scores -> softmax -> attn store -> ctx)
  attn_kernel<<<dim3(SEQ / 16, HEADS, BATCH), 32, 0, stream>>>(
      Qf, Kf, Vt, attn, Cx);
  // 3) output projection + bias
  out_proj_kernel<<<dim3(E_DIM / 64, NROWS / 256), 256, 0, stream>>>(
      Cx, WH + 3 * WE, bo, out);
}
